// MultiHeadLatentAttention_33509334843875
// MI455X (gfx1250) — compile-verified, matrix operand order checked by reading, not run on hardware
//
#include <hip/hip_runtime.h>

// ---------------------------------------------------------------------------
// MLA forward for MI455X (gfx1250, wave32, WMMA, async-LDS staging).
// Pipeline: convert operands to bf16 once; all GEMMs/attention run
// v_wmma_f32_16x16x32_bf16 with (async) LDS tiles, double-buffered in the GEMMs.
// ---------------------------------------------------------------------------

#define D_MODEL 2048
#define NHEADS  16
#define NOPE    128
#define ROPE    64
#define VDIM    128
#define QLORA   1536
#define KVLORA  512
#define QHD     (NOPE + ROPE)   // 192
#define SEQ     4096

typedef __attribute__((ext_vector_type(16))) __bf16       v16bf;
typedef __attribute__((ext_vector_type(8)))  float        v8f;
typedef __attribute__((ext_vector_type(4)))  unsigned int u32x4;
typedef __attribute__((ext_vector_type(4)))  int          i32x4;
typedef __attribute__((ext_vector_type(4)))  float        f32x4;

struct BF16x16 { unsigned short s[16]; };
struct U4x2    { u32x4 a, b; };

#if defined(__AMDGCN__) && __has_builtin(__builtin_amdgcn_global_load_async_to_lds_b128)
#define HAS_ASYNC 1
#else
#define HAS_ASYNC 0
#endif

// f32 -> bf16 (round-to-nearest-even)
__device__ __forceinline__ unsigned short f2bf(float f) {
    unsigned int u = __builtin_bit_cast(unsigned int, f);
    unsigned int r = u + 0x7FFFu + ((u >> 16) & 1u);
    return (unsigned short)(r >> 16);
}

__device__ __forceinline__ v8f wmma_bf16(v16bf a, v16bf b, v8f c) {
    return __builtin_amdgcn_wmma_f32_16x16x32_bf16(
        false, a, false, b, (short)0, c, false, false);
}

// Fragment (16 bf16/lane) from a bf16 row: elems 0..7 = K 8*hi.., 8..15 =
// K 16+8*hi..  Works for LDS and global pointers; rows padded to 16B align.
__device__ __forceinline__ v16bf frag16(const unsigned short* base, int hi) {
    U4x2 u;
    u.a = *(const u32x4*)(base + 8 * hi);
    u.b = *(const u32x4*)(base + 16 + 8 * hi);
    return __builtin_bit_cast(v16bf, u);
}

// 16-byte global->LDS copy: async (ASYNCcnt) when available, sync fallback.
__device__ __forceinline__ void async_copy16(const unsigned short* g,
                                             unsigned short* l) {
#if HAS_ASYNC
    __builtin_amdgcn_global_load_async_to_lds_b128(
        (__attribute__((address_space(1))) i32x4*)(g),
        (__attribute__((address_space(3))) i32x4*)(l), 0, 0);
#else
    *(u32x4*)l = *(const u32x4*)g;
#endif
}

__device__ __forceinline__ void wait_async() {
#if HAS_ASYNC
#if __has_builtin(__builtin_amdgcn_s_wait_asynccnt)
    __builtin_amdgcn_s_wait_asynccnt(0);
#else
    asm volatile("s_wait_asynccnt 0x0" ::: "memory");
#endif
#endif
}

// ---------------------------------------------------------------------------
// f32 -> bf16 bulk convert (n % 8 == 0)
// ---------------------------------------------------------------------------
__global__ __launch_bounds__(256) void f32_to_bf16_kernel(
    const float* __restrict__ in, unsigned short* __restrict__ out, int n) {
    int i = (blockIdx.x * 256 + threadIdx.x) * 8;
    if (i >= n) return;
    f32x4 a = *(const f32x4*)(in + i);
    f32x4 b = *(const f32x4*)(in + i + 4);
    u32x4 o;
    o.x = (unsigned)f2bf(a.x) | ((unsigned)f2bf(a.y) << 16);
    o.y = (unsigned)f2bf(a.z) | ((unsigned)f2bf(a.w) << 16);
    o.z = (unsigned)f2bf(b.x) | ((unsigned)f2bf(b.y) << 16);
    o.w = (unsigned)f2bf(b.z) | ((unsigned)f2bf(b.w) << 16);
    *(u32x4*)(out + i) = o;
}

// ---------------------------------------------------------------------------
// Transpose+convert weights: W (K x N) f32 -> WT (N x K) bf16.  K,N % 32 == 0.
// ---------------------------------------------------------------------------
__global__ __launch_bounds__(256) void transpose_bf16_kernel(
    const float* __restrict__ W, unsigned short* __restrict__ WT, int K, int N) {
    __shared__ unsigned short t[32][33];
    const int k0 = blockIdx.y * 32, n0 = blockIdx.x * 32;
#pragma unroll
    for (int i = 0; i < 4; i++) {
        int idx = threadIdx.x + i * 256;
        int kk = idx >> 5, nn = idx & 31;
        t[nn][kk] = f2bf(W[(unsigned)((k0 + kk) * N + n0 + nn)]);
    }
    __syncthreads();
#pragma unroll
    for (int i = 0; i < 4; i++) {
        int idx = threadIdx.x + i * 256;
        int nn = idx >> 5, kk = idx & 31;
        WT[(unsigned)((n0 + nn) * K + k0 + kk)] = t[nn][kk];
    }
}

// ---------------------------------------------------------------------------
// GEMM: C(M,N) = A(M,K) @ BT(N,K)^T, bf16 in (A row-major, B transposed),
// f32 or bf16 out.  128x128 block tile, 8 waves, 32x64 wave tile.
// Double-buffered LDS fed by per-lane async B128 copies.
// M % 128 == 0, K % 32 == 0; BT rows must be padded to a multiple of 128
// (garbage pad rows are masked by the col<N store guard).
// ---------------------------------------------------------------------------
template <bool OUT_BF16>
__global__ __launch_bounds__(256) void gemm_bf16_kernel(
    const unsigned short* __restrict__ A, const unsigned short* __restrict__ BT,
    void* __restrict__ Cv, int M, int N, int K, int lda, int ldbt, int ldc) {
    __shared__ __align__(16) unsigned short lA[2][128 * 40];  // [m][k] pad 40
    __shared__ __align__(16) unsigned short lB[2][128 * 40];  // [n][k] pad 40

    const int tid  = threadIdx.x;
    const int lane = tid & 31;
    const int ln   = lane & 15;
    const int hi   = lane >> 4;
    const int wm   = (tid >> 5) & 3;   // 4 row groups of 32
    const int wn   = tid >> 7;         // 2 col groups of 64
    const int m0   = blockIdx.y * 128;
    const int n0   = blockIdx.x * 128;

    v8f acc[2][4];
#pragma unroll
    for (int mi = 0; mi < 2; mi++)
#pragma unroll
        for (int ni = 0; ni < 4; ni++)
            acc[mi][ni] = (v8f){0.f, 0.f, 0.f, 0.f, 0.f, 0.f, 0.f, 0.f};

    auto stage = [&](int buf, int k0) {
#pragma unroll
        for (int i = 0; i < 2; i++) {
            int c0 = tid + i * 256;                 // 512 chunks of 16B each
            int row = c0 >> 2, seg = (c0 & 3) * 8;
            async_copy16(A + (unsigned)((m0 + row) * lda + k0) + seg,
                         &lA[buf][row * 40 + seg]);
            async_copy16(BT + (unsigned)((n0 + row) * ldbt + k0) + seg,
                         &lB[buf][row * 40 + seg]);
        }
    };

    stage(0, 0);
    for (int k0 = 0; k0 < K; k0 += 32) {
        const int cur = (k0 >> 5) & 1;
        wait_async();          // my async stores into buf `cur` have landed
        __syncthreads();       // everyone's have; everyone done reading cur^1
        if (k0 + 32 < K) stage(cur ^ 1, k0 + 32);

        v16bf af[2], bfr[4];
#pragma unroll
        for (int mi = 0; mi < 2; mi++)
            af[mi] = frag16(&lA[cur][(wm * 32 + mi * 16 + ln) * 40], hi);
#pragma unroll
        for (int ni = 0; ni < 4; ni++)
            bfr[ni] = frag16(&lB[cur][(wn * 64 + ni * 16 + ln) * 40], hi);
#pragma unroll
        for (int mi = 0; mi < 2; mi++)
#pragma unroll
            for (int ni = 0; ni < 4; ni++)
                acc[mi][ni] = wmma_bf16(af[mi], bfr[ni], acc[mi][ni]);
        __syncthreads();       // reads of buf `cur` done before it is re-staged
    }

#pragma unroll
    for (int mi = 0; mi < 2; mi++)
#pragma unroll
        for (int ni = 0; ni < 4; ni++) {
            int col = n0 + wn * 64 + ni * 16 + ln;
            if (col < N) {
#pragma unroll
                for (int r = 0; r < 8; r++) {
                    unsigned idx =
                        (unsigned)((m0 + wm * 32 + mi * 16 + r + 8 * hi) * ldc + col);
                    if (OUT_BF16)
                        ((unsigned short*)Cv)[idx] = f2bf(acc[mi][ni][r]);
                    else
                        ((float*)Cv)[idx] = acc[mi][ni][r];
                }
            }
        }
}

// ---------------------------------------------------------------------------
// RMSNorm (f32 in, bf16 out): out = in * rsqrt(mean(in^2)+eps) * w
// ---------------------------------------------------------------------------
__global__ __launch_bounds__(256) void rmsnorm_kernel(
    const float* __restrict__ in, const float* __restrict__ w,
    unsigned short* __restrict__ out, int L, int ldin, int ldout) {
    const int row = blockIdx.x;
    const float* ip = in + (unsigned)(row * ldin);
    float ss = 0.f;
    for (int i = threadIdx.x; i < L; i += 256) { float v = ip[i]; ss += v * v; }
#pragma unroll
    for (int off = 16; off > 0; off >>= 1) ss += __shfl_xor(ss, off);
    __shared__ float red[8];
    __shared__ float stot;
    if ((threadIdx.x & 31) == 0) red[threadIdx.x >> 5] = ss;
    __syncthreads();
    if (threadIdx.x == 0) {
        float t = 0.f;
#pragma unroll
        for (int i = 0; i < 8; i++) t += red[i];
        stot = t / (float)L;
    }
    __syncthreads();
    float scale = rsqrtf(stot + 1.1920929e-07f);
    unsigned short* op = out + (unsigned)(row * ldout);
    for (int i = threadIdx.x; i < L; i += 256) op[i] = f2bf(ip[i] * scale * w[i]);
}

// ---------------------------------------------------------------------------
// RoPE on q (f32, in place): roped dims are [128,192) of each head.
// ---------------------------------------------------------------------------
__global__ __launch_bounds__(256) void rope_q_kernel(float* __restrict__ q) {
    int idx = blockIdx.x * 256 + threadIdx.x;     // SEQ*NHEADS*32 pairs
    if (idx >= SEQ * NHEADS * 32) return;
    int p = idx & 31, h = (idx >> 5) & 15, t = idx >> 9;
    float freq = __expf(-(float)(2 * p) * (9.210340371976184f / 64.f));
    float ang = (float)t * freq;
    float c, s;
    __sincosf(ang, &c, &s);
    float* b = q + (unsigned)(t * (NHEADS * QHD) + h * QHD + NOPE + 2 * p);
    float xe = b[0], xo = b[1];
    b[0] = xe * c - xo * s;
    b[1] = xo * c + xe * s;
}

// RoPE on k_rope: kv f32 (ld 576) cols [512,576) -> kr bf16 (SEQ,64).
__global__ __launch_bounds__(256) void rope_k_kernel(
    const float* __restrict__ kv, unsigned short* __restrict__ kr) {
    int idx = blockIdx.x * 256 + threadIdx.x;     // SEQ*32 pairs
    if (idx >= SEQ * 32) return;
    int p = idx & 31, t = idx >> 5;
    float freq = __expf(-(float)(2 * p) * (9.210340371976184f / 64.f));
    float ang = (float)t * freq;
    float c, s;
    __sincosf(ang, &c, &s);
    const float* src = kv + (unsigned)(t * (KVLORA + ROPE) + KVLORA + 2 * p);
    float xe = src[0], xo = src[1];
    kr[(unsigned)(t * ROPE + 2 * p)]     = f2bf(xe * c - xo * s);
    kr[(unsigned)(t * ROPE + 2 * p + 1)] = f2bf(xo * c + xe * s);
}

// ---------------------------------------------------------------------------
// Flash MLA attention, all-bf16 operands.  Grid (SEQ/128, NHEADS), 8 waves.
// Per 32-key block: 12 score WMMAs + 8 PV WMMAs.  Causal loop bound.
// q:(SEQ,3072) roped; kvb:(SEQ,4096) [k_nope|v] per head; kr:(SEQ,64);
// o:(SEQ,2048) bf16.
// ---------------------------------------------------------------------------
__global__ __launch_bounds__(256) void mla_attn_kernel(
    const unsigned short* __restrict__ q, const unsigned short* __restrict__ kvb,
    const unsigned short* __restrict__ kr, unsigned short* __restrict__ obuf) {
    const int h  = blockIdx.y;
    const int qb = blockIdx.x;
    const int tid = threadIdx.x;
    const int w = tid >> 5, lane = tid & 31;
    const int ln = lane & 15, hi = lane >> 4;
    const int qrow0 = qb * 128 + w * 16;

    __shared__ __align__(16) unsigned short lK[32 * 200];   // [key][d] pad 200
    __shared__ __align__(16) unsigned short lV[128 * 40];   // [vd][key] pad 40
    __shared__ __align__(16) unsigned short lP[8][16 * 40]; // per-wave P

    // Q fragments straight from bf16 global: 6 K-chunks of 32 (d = 0..191).
    v16bf qf[6];
    {
        const unsigned short* qp =
            q + (unsigned)((qrow0 + ln) * (NHEADS * QHD) + h * QHD);
#pragma unroll
        for (int c = 0; c < 6; c++) qf[c] = frag16(qp + c * 32, hi);
    }

    v8f o[8];
#pragma unroll
    for (int t = 0; t < 8; t++) o[t] = (v8f){0.f, 0.f, 0.f, 0.f, 0.f, 0.f, 0.f, 0.f};
    float mrow[8], lrow[8];
#pragma unroll
    for (int r = 0; r < 8; r++) { mrow[r] = -1e30f; lrow[r] = 0.f; }

    const float scale = 0.07216878364870323f;  // 1/sqrt(192)
    const int nkb = qb * 4 + 4;

    for (int kb = 0; kb < nkb; kb++) {
        const int k0 = kb * 32;
        __syncthreads();
        // K tile (32 x 192 bf16): pure byte copy -> async B128 per lane.
#pragma unroll
        for (int i = 0; i < 3; i++) {
            int c0 = tid + i * 256;                // 768 chunks of 16B
            int key = c0 / 24, seg = (c0 - key * 24) * 8;
            const unsigned short* src =
                (seg < NOPE)
                    ? kvb + (unsigned)((k0 + key) * (NHEADS * 256) + h * 256 + seg)
                    : kr + (unsigned)((k0 + key) * ROPE + (seg - NOPE));
            async_copy16(src, &lK[key * 200 + seg]);
        }
        // V^T tile: bf16 pairs along vd, scattered into [vd][key].
#pragma unroll
        for (int i = 0; i < 8; i++) {
            int c0 = tid + i * 256;                // 2048 pairs
            int key = c0 >> 6, vp = (c0 & 63) * 2;
            unsigned int d = *(const unsigned int*)(
                kvb + (unsigned)((k0 + key) * (NHEADS * 256) + h * 256 + NOPE + vp));
            lV[vp * 40 + key]       = (unsigned short)(d & 0xFFFFu);
            lV[(vp + 1) * 40 + key] = (unsigned short)(d >> 16);
        }
        wait_async();
        __syncthreads();

        // Scores: S(16q x 32k) as two 16x16 tiles.
        v8f s0 = (v8f){0.f, 0.f, 0.f, 0.f, 0.f, 0.f, 0.f, 0.f};
        v8f s1 = s0;
#pragma unroll
        for (int c = 0; c < 6; c++) {
            v16bf kf0 = frag16(&lK[ln * 200 + c * 32], hi);
            v16bf kf1 = frag16(&lK[(16 + ln) * 200 + c * 32], hi);
            s0 = wmma_bf16(qf[c], kf0, s0);
            s1 = wmma_bf16(qf[c], kf1, s1);
        }

        // Online softmax (row stats across lanes 0..15 of each half-wave).
#pragma unroll
        for (int r = 0; r < 8; r++) {
            int tq = qrow0 + r + 8 * hi;
            float a0 = (k0 + ln      <= tq) ? s0[r] * scale : -1e30f;
            float a1 = (k0 + 16 + ln <= tq) ? s1[r] * scale : -1e30f;
            float mt = fmaxf(a0, a1);
#pragma unroll
            for (int msk = 1; msk < 16; msk <<= 1) mt = fmaxf(mt, __shfl_xor(mt, msk));
            float mnew  = fmaxf(mrow[r], mt);
            float alpha = __expf(mrow[r] - mnew);
            float p0 = __expf(a0 - mnew);
            float p1 = __expf(a1 - mnew);
            float rs = p0 + p1;
#pragma unroll
            for (int msk = 1; msk < 16; msk <<= 1) rs += __shfl_xor(rs, msk);
            lrow[r] = lrow[r] * alpha + rs;
            mrow[r] = mnew;
#pragma unroll
            for (int t = 0; t < 8; t++) o[t][r] *= alpha;
            lP[w][(r + 8 * hi) * 40 + ln]      = f2bf(p0);
            lP[w][(r + 8 * hi) * 40 + 16 + ln] = f2bf(p1);
        }

        // P(16x32) @ V(32x128): 8 WMMAs.
        v16bf pf = frag16(&lP[w][ln * 40], hi);
#pragma unroll
        for (int t = 0; t < 8; t++) {
            v16bf vf = frag16(&lV[(t * 16 + ln) * 40], hi);
            o[t] = wmma_bf16(pf, vf, o[t]);
        }
    }

#pragma unroll
    for (int r = 0; r < 8; r++) {
        float inv = 1.f / lrow[r];
        int row = qrow0 + r + 8 * hi;
#pragma unroll
        for (int t = 0; t < 8; t++)
            obuf[(unsigned)(row * (NHEADS * VDIM) + h * VDIM + t * 16 + ln)] =
                f2bf(o[t][r] * inv);
    }
}

// ---------------------------------------------------------------------------
extern "C" void kernel_launch(void* const* d_in, const int* in_sizes, int n_in,
                              void* d_out, int out_size, void* d_ws, size_t ws_size,
                              hipStream_t stream) {
    (void)in_sizes; (void)n_in; (void)out_size; (void)ws_size;
    const float* x       = (const float*)d_in[0];
    const float* Wqa     = (const float*)d_in[1];
    const float* q_ln_w  = (const float*)d_in[2];
    const float* Wqb     = (const float*)d_in[3];
    const float* Wkva    = (const float*)d_in[4];
    const float* kv_ln_w = (const float*)d_in[5];
    const float* Wkvb    = (const float*)d_in[6];
    const float* Wo      = (const float*)d_in[7];
    float* out = (float*)d_out;
    char*  ws  = (char*)d_ws;

    // Workspace layout (bytes). Total 169,345,024.
    unsigned short* xbf   = (unsigned short*)(ws);              // 8.4M el
    unsigned short* wT    = (unsigned short*)(ws + 16777216);   // <=4.72M el (reused)
    unsigned short* qbf   = (unsigned short*)(ws + 26214400);   // SEQ*3072
    unsigned short* qlatb = (unsigned short*)(ws + 51380224);   // SEQ*1536
    unsigned short* kvcb  = (unsigned short*)(ws + 63963136);   // SEQ*512
    unsigned short* krbf  = (unsigned short*)(ws + 68157440);   // SEQ*64
    unsigned short* kvbbf = (unsigned short*)(ws + 68681728);   // SEQ*4096
    unsigned short* obf   = (unsigned short*)(ws + 102236160);  // SEQ*2048
    float* qbuf  = (float*)(ws + 119013376);                    // SEQ*3072 f32
    float* qlatf = (float*)(ws + 68681728);   // aliases kvbbf (dead before GEMM4)
    float* kvbf  = (float*)(ws + 102236160);  // aliases obf   (dead before attn)

    dim3 blk(256);

    // x -> bf16
    f32_to_bf16_kernel<<<SEQ * D_MODEL / 8 / 256, blk, 0, stream>>>(
        x, xbf, SEQ * D_MODEL);
    // q_lat = x @ Wqa
    transpose_bf16_kernel<<<dim3(QLORA / 32, D_MODEL / 32), blk, 0, stream>>>(
        Wqa, wT, D_MODEL, QLORA);
    gemm_bf16_kernel<false><<<dim3(QLORA / 128, SEQ / 128), blk, 0, stream>>>(
        xbf, wT, qlatf, SEQ, QLORA, D_MODEL, D_MODEL, D_MODEL, QLORA);
    rmsnorm_kernel<<<SEQ, blk, 0, stream>>>(qlatf, q_ln_w, qlatb, QLORA, QLORA, QLORA);
    // q = rms(q_lat) @ Wqb
    transpose_bf16_kernel<<<dim3(NHEADS * QHD / 32, QLORA / 32), blk, 0, stream>>>(
        Wqb, wT, QLORA, NHEADS * QHD);
    gemm_bf16_kernel<false><<<dim3(NHEADS * QHD / 128, SEQ / 128), blk, 0, stream>>>(
        qlatb, wT, qbuf, SEQ, NHEADS * QHD, QLORA, QLORA, QLORA, NHEADS * QHD);
    rope_q_kernel<<<SEQ * NHEADS * 32 / 256, blk, 0, stream>>>(qbuf);
    f32_to_bf16_kernel<<<SEQ * NHEADS * QHD / 8 / 256, blk, 0, stream>>>(
        qbuf, qbf, SEQ * NHEADS * QHD);
    // kv = x @ Wkva
    transpose_bf16_kernel<<<dim3((KVLORA + ROPE) / 32, D_MODEL / 32), blk, 0, stream>>>(
        Wkva, wT, D_MODEL, KVLORA + ROPE);
    gemm_bf16_kernel<false><<<dim3((KVLORA + ROPE + 127) / 128, SEQ / 128), blk, 0, stream>>>(
        xbf, wT, kvbf, SEQ, KVLORA + ROPE, D_MODEL, D_MODEL, D_MODEL, KVLORA + ROPE);
    rmsnorm_kernel<<<SEQ, blk, 0, stream>>>(kvbf, kv_ln_w, kvcb, KVLORA,
                                            KVLORA + ROPE, KVLORA);
    rope_k_kernel<<<SEQ * 32 / 256, blk, 0, stream>>>(kvbf, krbf);
    // kvb = rms(kv_c) @ Wkvb   (bf16 out)
    transpose_bf16_kernel<<<dim3(NHEADS * 256 / 32, KVLORA / 32), blk, 0, stream>>>(
        Wkvb, wT, KVLORA, NHEADS * 256);
    gemm_bf16_kernel<true><<<dim3(NHEADS * 256 / 128, SEQ / 128), blk, 0, stream>>>(
        kvcb, wT, kvbbf, SEQ, NHEADS * 256, KVLORA, KVLORA, KVLORA, NHEADS * 256);
    // attention
    mla_attn_kernel<<<dim3(SEQ / 128, NHEADS), blk, 0, stream>>>(qbf, kvbbf, krbf, obf);
    // out = o @ Wo
    transpose_bf16_kernel<<<dim3(D_MODEL / 32, D_MODEL / 32), blk, 0, stream>>>(
        Wo, wT, NHEADS * VDIM, D_MODEL);
    gemm_bf16_kernel<false><<<dim3(D_MODEL / 128, SEQ / 128), blk, 0, stream>>>(
        obf, wT, out, SEQ, D_MODEL, NHEADS * VDIM, NHEADS * VDIM, NHEADS * VDIM, D_MODEL);
}
